// BATPoseDecoder_7894149890256
// MI455X (gfx1250) — matrix operand order repaired and checked by reading, not compiled, not run on hardware
//
#include <hip/hip_runtime.h>

typedef unsigned short u16;
typedef unsigned int   u32;
typedef unsigned long long u64;

#define D_DIM 524
#define KP    544   // K padded to multiple of 32 for GEMM1 (17 chunks)
#define NPTS  256
#define MPTS  256

typedef __attribute__((ext_vector_type(16))) __bf16 v16bf;
typedef __attribute__((ext_vector_type(8)))  float  v8f;

struct Bits256 { u32 u[8]; };

__device__ inline u16 f2bf(float f) {
  u32 u = __float_as_uint(f);
  u32 r = u + 0x7FFFu + ((u >> 16) & 1u);   // RNE
  return (u16)(r >> 16);
}
__device__ inline u32 pk2(float lo, float hi) {
  return (u32)f2bf(lo) | ((u32)f2bf(hi) << 16);
}
__device__ inline v16bf frag(const u16* p0, const u16* p1) {
  uint4 a = *reinterpret_cast<const uint4*>(p0);
  uint4 b = *reinterpret_cast<const uint4*>(p1);
  Bits256 t;
  t.u[0]=a.x; t.u[1]=a.y; t.u[2]=a.z; t.u[3]=a.w;
  t.u[4]=b.x; t.u[5]=b.y; t.u[6]=b.z; t.u[7]=b.w;
  return __builtin_bit_cast(v16bf, t);
}
__device__ inline v8f zero8() { v8f z = {0.f,0.f,0.f,0.f,0.f,0.f,0.f,0.f}; return z; }

// ---- CDNA5 async global->LDS DMA (ASYNCcnt-tracked) ----
__device__ __forceinline__ void async_lds_b128(u32 lds_off, u64 gaddr) {
  asm volatile("global_load_async_to_lds_b128 %0, %1, off"
               :: "v"(lds_off), "v"(gaddr)
               : "memory");
}
__device__ __forceinline__ void wait_async0() {
  asm volatile("s_wait_asynccnt 0x0" ::: "memory");
}

// ---------------- weight conversion ----------------
__global__ void conv_w1_kernel(const float* __restrict__ cw1, u16* __restrict__ W1bf) {
  int i = blockIdx.x * 256 + threadIdx.x;       // [544][256]
  if (i >= KP * 256) return;
  int k = i >> 8;
  float v = (k < D_DIM) ? cw1[i] : 0.f;         // cw1 is [524][256] row-major
  W1bf[i] = f2bf(v);
}
__global__ void conv_w2_kernel(const float* __restrict__ cw2, u16* __restrict__ W2bf) {
  int i = blockIdx.x * 256 + threadIdx.x;       // [256][128]
  if (i >= 256 * 128) return;
  W2bf[i] = f2bf(cw2[i]);
}

// ---- staging helpers ----
template<bool TAIL>
__device__ __forceinline__ void stageA(const float* __restrict__ X, int mb, int kk,
                                       u16* __restrict__ dst, int tid) {
  int r = tid >> 3, seg = tid & 7;
  int kg = kk + seg * 4;
  const float* xr = X + (size_t)(mb * 32 + r) * D_DIM;
  float f0, f1, f2, f3;
  if (!TAIL || (kg + 4 <= D_DIM)) {
    float4 v = *reinterpret_cast<const float4*>(xr + kg);
    f0 = v.x; f1 = v.y; f2 = v.z; f3 = v.w;
  } else {
    f0 = (kg + 0 < D_DIM) ? xr[kg + 0] : 0.f;
    f1 = (kg + 1 < D_DIM) ? xr[kg + 1] : 0.f;
    f2 = (kg + 2 < D_DIM) ? xr[kg + 2] : 0.f;
    f3 = (kg + 3 < D_DIM) ? xr[kg + 3] : 0.f;
  }
  uint2 pv; pv.x = pk2(f0, f1); pv.y = pk2(f2, f3);
  *reinterpret_cast<uint2*>(&dst[r * 40 + seg * 4]) = pv;
}

__device__ __forceinline__ void stageB1(const u16* __restrict__ W1bf, int kk,
                                        u16* __restrict__ dstBase, int tid) {
  int kr = tid >> 3, seg = tid & 7;                       // 32 rows x 256 cols
  u64 g = (u64)(size_t)(W1bf + (size_t)(kk + kr) * 256 + seg * 32);
  u32 off = (u32)(size_t)(dstBase + kr * 264 + seg * 32);
  #pragma unroll
  for (int i = 0; i < 4; ++i) async_lds_b128(off + i * 16, g + i * 16);
}
__device__ __forceinline__ void stageB2(const u16* __restrict__ W2bf, int kk,
                                        u16* __restrict__ dstBase, int tid) {
  int kr = tid >> 3, seg = tid & 7;                       // 32 rows x 128 cols
  u64 g = (u64)(size_t)(W2bf + (size_t)(kk + kr) * 128 + seg * 16);
  u32 off = (u32)(size_t)(dstBase + kr * 264 + seg * 16);
  #pragma unroll
  for (int i = 0; i < 2; ++i) async_lds_b128(off + i * 16, g + i * 16);
}

// ---------------- fused corr MLP + softmax + matched ----------------
__global__ __launch_bounds__(256) void corr_kernel(
    const float* __restrict__ Lbat, const float* __restrict__ tgt_cent,
    const float* __restrict__ cb1,  const float* __restrict__ cb2,
    const float* __restrict__ cw3,  const float* __restrict__ cb3,
    const u16*  __restrict__ W1bf,  const u16*  __restrict__ W2bf,
    float* __restrict__ corr_out,   float* __restrict__ matched_ws,
    float* __restrict__ tgtm_ws)
{
  __shared__ __align__(16) u16   sA[2][32 * 40];   // X chunk, bf16, double buffered
  __shared__ __align__(16) u16   sB[2][32 * 264];  // weight chunk [k][n], double buffered
  __shared__ __align__(16) u16   sH1[32 * 264];    // h1 block bf16 (reused as fp32 h2)
  __shared__ float sScores[256];
  __shared__ float sRed[256];
  __shared__ float sCw3[128];

  const int bn   = blockIdx.x;
  const int b    = bn >> 8;
  const int tid  = threadIdx.x;
  const int w    = tid >> 5;
  const int lane = tid & 31;
  const int l    = lane & 15;
  const int hi16 = (lane >> 4) & 1;

  const float* X = Lbat + (size_t)bn * MPTS * D_DIM;

  if (tid < 128) sCw3[tid] = cw3[tid];

  // GEMM1: out 32x256 -> 2 m-tiles x 16 n-tiles; wave: 1 m-tile, 4 n-tiles
  const int mt1 = w >> 2;
  const int nb1 = (w & 3) * 4;
  // GEMM2: out 32x128 -> 2 m-tiles x 8 n-tiles; wave: 1 m-tile, 2 n-tiles
  const int nb2 = (w & 3) * 2;

  const int arow  = mt1 * 16 + l;
  const int kA0   = hi16 ? 8 : 0;        // A-frag K split per ISA layout
  const int krowB = hi16 * 16 + l;       // unused; B-frag lane = N column, k in elements

  float* hf = reinterpret_cast<float*>(sH1);   // fp32 h2 view, stride 132

  for (int mb = 0; mb < 8; ++mb) {
    // -------- GEMM1: h1[32,256] = X[32,Kp] @ W1 (pipelined, async weights) ----
    v8f acc[4];
    #pragma unroll
    for (int i = 0; i < 4; ++i) acc[i] = zero8();

    stageA<false>(X, mb, 0, sA[0], tid);
    stageB1(W1bf, 0, sB[0], tid);
    wait_async0();
    __syncthreads();

    for (int ic = 0; ic < 17; ++ic) {
      const int p = ic & 1;
      // prefetch next chunk into the other buffer
      if (ic + 1 < 17) {
        int kn = (ic + 1) * 32;
        if (ic + 1 == 16) stageA<true>(X, mb, kn, sA[p ^ 1], tid);
        else              stageA<false>(X, mb, kn, sA[p ^ 1], tid);
        stageB1(W1bf, kn, sB[p ^ 1], tid);
      }
      const u16* aB = sA[p];
      const u16* bB = sB[p];
      v16bf af = frag(&aB[arow * 40 + kA0], &aB[arow * 40 + kA0 + 16]);
      #pragma unroll
      for (int i = 0; i < 4; ++i) {
        int nc = (nb1 + i) * 16;
        v16bf bfr;
        #pragma unroll
        for (int s = 0; s < 16; ++s)
          bfr[s] = __builtin_bit_cast(__bf16, bB[(kA0 + (s & 7) + (s >> 3) * 16) * 264 + nc + l]);
        acc[i] = __builtin_amdgcn_wmma_f32_16x16x32_bf16(
            false, af, false, bfr, (short)0, acc[i], false, false);
      }
      wait_async0();
      __syncthreads();
    }
    { // epilogue: bias + relu -> sH1 bf16
      int mr = mt1 * 16 + hi16 * 8;
      #pragma unroll
      for (int i = 0; i < 4; ++i) {
        int n = (nb1 + i) * 16 + l;
        float bias = cb1[n];
        #pragma unroll
        for (int j = 0; j < 8; ++j) {
          float v = fmaxf(acc[i][j] + bias, 0.f);
          sH1[(mr + j) * 264 + n] = f2bf(v);
        }
      }
    }

    // -------- GEMM2: h2[32,128] = h1[32,256] @ W2 (pipelined) --------
    v8f acc2[2];
    #pragma unroll
    for (int i = 0; i < 2; ++i) acc2[i] = zero8();

    stageB2(W2bf, 0, sB[0], tid);
    wait_async0();
    __syncthreads();   // also publishes sH1 epilogue writes

    for (int ic = 0; ic < 8; ++ic) {
      const int p = ic & 1;
      if (ic + 1 < 8) stageB2(W2bf, (ic + 1) * 32, sB[p ^ 1], tid);
      int kk = ic * 32;
      const u16* bB = sB[p];
      v16bf af = frag(&sH1[arow * 264 + kk + kA0], &sH1[arow * 264 + kk + kA0 + 16]);
      #pragma unroll
      for (int i = 0; i < 2; ++i) {
        int nc = (nb2 + i) * 16;
        v16bf bfr;
        #pragma unroll
        for (int s = 0; s < 16; ++s)
          bfr[s] = __builtin_bit_cast(__bf16, bB[(kA0 + (s & 7) + (s >> 3) * 16) * 264 + nc + l]);
        acc2[i] = __builtin_amdgcn_wmma_f32_16x16x32_bf16(
            false, af, false, bfr, (short)0, acc2[i], false, false);
      }
      wait_async0();
      __syncthreads();
    }
    { // epilogue: bias + relu -> fp32 h2 in LDS (overwrites h1, already consumed)
      int mr = mt1 * 16 + hi16 * 8;
      #pragma unroll
      for (int i = 0; i < 2; ++i) {
        int n = (nb2 + i) * 16 + l;
        float bias = cb2[n];
        #pragma unroll
        for (int j = 0; j < 8; ++j)
          hf[(mr + j) * 132 + n] = fmaxf(acc2[i][j] + bias, 0.f);
      }
    }
    __syncthreads();
    // scores[m] = h2[m,:] . cw3 + cb3 (deterministic, per-row)
    if (tid < 32) {
      float s = cb3[0];
      const float* hr = hf + tid * 132;
      #pragma unroll 4
      for (int n = 0; n < 128; ++n) s += hr[n] * sCw3[n];
      sScores[mb * 32 + tid] = s;
    }
    __syncthreads();
  }

  // -------- softmax over M=256 --------
  float sc = sScores[tid];
  sRed[tid] = sc; __syncthreads();
  for (int off = 128; off > 0; off >>= 1) {
    if (tid < off) sRed[tid] = fmaxf(sRed[tid], sRed[tid + off]);
    __syncthreads();
  }
  float mx = sRed[0]; __syncthreads();
  float e = __expf(sc - mx);
  sRed[tid] = e; __syncthreads();
  for (int off = 128; off > 0; off >>= 1) {
    if (tid < off) sRed[tid] += sRed[tid + off];
    __syncthreads();
  }
  float inv = 1.f / sRed[0];
  __syncthreads();
  float wgt = e * inv;
  sScores[tid] = wgt;
  corr_out[(size_t)bn * MPTS + tid] = wgt;
  __syncthreads();

  // -------- matched[b,n,:] = weights . X  (coalesced over d) --------
  #pragma unroll
  for (int j = 0; j < 3; ++j) {
    int d = tid + j * 256;
    if (d < D_DIM) {
      float a = 0.f;
      #pragma unroll 4
      for (int m = 0; m < MPTS; ++m) a += sScores[m] * X[(size_t)m * D_DIM + d];
      matched_ws[(size_t)bn * D_DIM + d] = a;
    }
  }
  if (tid < 3) {
    float a = 0.f;
    for (int m = 0; m < MPTS; ++m)
      a += sScores[m] * tgt_cent[((size_t)b * MPTS + m) * 3 + tid];
    tgtm_ws[(size_t)bn * 3 + tid] = a;
  }
}

// ---------------- overlap MLP (tiny, VALU) ----------------
__global__ __launch_bounds__(128) void overlap_kernel(
    const float* __restrict__ matched, const float* __restrict__ ow1,
    const float* __restrict__ ob1, const float* __restrict__ ow2,
    const float* __restrict__ ob2, float* __restrict__ overlap_out)
{
  __shared__ float sM[D_DIM];
  __shared__ float sR[128];
  int bn = blockIdx.x, t = threadIdx.x;
  const float* mrow = matched + (size_t)bn * D_DIM;
  for (int d = t; d < D_DIM; d += 128) sM[d] = mrow[d];
  __syncthreads();
  float a = ob1[t];
  #pragma unroll 4
  for (int d = 0; d < D_DIM; ++d) a = fmaf(sM[d], ow1[d * 128 + t], a);
  a = fmaxf(a, 0.f);
  sR[t] = a * ow2[t];
  __syncthreads();
  for (int off = 64; off > 0; off >>= 1) {
    if (t < off) sR[t] += sR[t + off];
    __syncthreads();
  }
  if (t == 0) overlap_out[bn] = 1.f / (1.f + __expf(-(sR[0] + ob2[0])));
}

// ---------------- weighted Procrustes + quaternion ----------------
__device__ inline float blk_sum(float* sRed, int t, float v) {
  sRed[t] = v; __syncthreads();
  for (int off = 128; off > 0; off >>= 1) {
    if (t < off) sRed[t] += sRed[t + off];
    __syncthreads();
  }
  float r = sRed[0]; __syncthreads();
  return r;
}

__device__ inline float det3(const float M[3][3]) {
  return M[0][0] * (M[1][1] * M[2][2] - M[1][2] * M[2][1])
       - M[0][1] * (M[1][0] * M[2][2] - M[1][2] * M[2][0])
       + M[0][2] * (M[1][0] * M[2][1] - M[1][1] * M[2][0]);
}

__global__ __launch_bounds__(256) void pose_kernel(
    const float* __restrict__ src_cent, const float* __restrict__ tgtm_ws,
    const float* __restrict__ overlap, float* __restrict__ quat_out,
    float* __restrict__ trans_out, float* __restrict__ R_out)
{
  __shared__ float sRed[256];
  int b = blockIdx.x, t = threadIdx.x;
  float wv = overlap[b * 256 + t];
  float wsum = blk_sum(sRed, t, wv);
  float wn = wv / fmaxf(wsum, 1e-8f);

  float s[3], g[3];
  #pragma unroll
  for (int c = 0; c < 3; ++c) {
    s[c] = src_cent[((size_t)b * 256 + t) * 3 + c];
    g[c] = tgtm_ws[((size_t)b * 256 + t) * 3 + c];
  }
  float sb[3], gb[3];
  for (int c = 0; c < 3; ++c) sb[c] = blk_sum(sRed, t, wn * s[c]);
  for (int c = 0; c < 3; ++c) gb[c] = blk_sum(sRed, t, wn * g[c]);
  float scv[3], tcv[3];
  #pragma unroll
  for (int c = 0; c < 3; ++c) { scv[c] = s[c] - sb[c]; tcv[c] = g[c] - gb[c]; }
  float Hm[3][3];
  for (int c = 0; c < 3; ++c)
    for (int d = 0; d < 3; ++d)
      Hm[c][d] = blk_sum(sRed, t, wn * scv[c] * tcv[d]);

  if (t == 0) {
    // A = H^T H, Jacobi eigendecomposition -> V, eigvals
    float A[3][3], V[3][3];
    for (int i = 0; i < 3; ++i)
      for (int j = 0; j < 3; ++j) {
        float acc = 0.f;
        for (int k = 0; k < 3; ++k) acc += Hm[k][i] * Hm[k][j];
        A[i][j] = acc;
        V[i][j] = (i == j) ? 1.f : 0.f;
      }
    const int prs[3][2] = {{0, 1}, {0, 2}, {1, 2}};
    for (int sweep = 0; sweep < 12; ++sweep) {
      for (int pi = 0; pi < 3; ++pi) {
        int p = prs[pi][0], q = prs[pi][1];
        float apq = A[p][q];
        if (fabsf(apq) < 1e-15f) continue;
        float th = (A[q][q] - A[p][p]) / (2.f * apq);
        float tt = ((th >= 0.f) ? 1.f : -1.f) / (fabsf(th) + sqrtf(th * th + 1.f));
        float c = 1.f / sqrtf(tt * tt + 1.f);
        float sn = tt * c;
        for (int k = 0; k < 3; ++k) {
          float akp = A[k][p], akq = A[k][q];
          A[k][p] = c * akp - sn * akq;
          A[k][q] = sn * akp + c * akq;
        }
        for (int k = 0; k < 3; ++k) {
          float apk = A[p][k], aqk = A[q][k];
          A[p][k] = c * apk - sn * aqk;
          A[q][k] = sn * apk + c * aqk;
        }
        for (int k = 0; k < 3; ++k) {
          float vkp = V[k][p], vkq = V[k][q];
          V[k][p] = c * vkp - sn * vkq;
          V[k][q] = sn * vkp + c * vkq;
        }
      }
    }
    float ev[3] = {A[0][0], A[1][1], A[2][2]};
    for (int i = 0; i < 2; ++i)
      for (int j = 0; j < 2 - i; ++j)
        if (ev[j] < ev[j + 1]) {
          float te = ev[j]; ev[j] = ev[j + 1]; ev[j + 1] = te;
          for (int k = 0; k < 3; ++k) {
            float tv = V[k][j]; V[k][j] = V[k][j + 1]; V[k][j + 1] = tv;
          }
        }
    float Sv[3], U[3][3];
    for (int i = 0; i < 3; ++i) Sv[i] = sqrtf(fmaxf(ev[i], 0.f));
    for (int i = 0; i < 3; ++i) {
      float u0 = 0.f, u1 = 0.f, u2 = 0.f;
      for (int k = 0; k < 3; ++k) {
        u0 += Hm[0][k] * V[k][i];
        u1 += Hm[1][k] * V[k][i];
        u2 += Hm[2][k] * V[k][i];
      }
      float nn = fmaxf(Sv[i], 1e-12f);
      U[0][i] = u0 / nn; U[1][i] = u1 / nn; U[2][i] = u2 / nn;
    }
    if (Sv[2] < 1e-6f * fmaxf(Sv[0], 1e-12f)) { // rank-deficient fallback
      float cx = U[1][0] * U[2][1] - U[2][0] * U[1][1];
      float cy = U[2][0] * U[0][1] - U[0][0] * U[2][1];
      float cz = U[0][0] * U[1][1] - U[1][0] * U[0][1];
      float nn = fmaxf(sqrtf(cx * cx + cy * cy + cz * cz), 1e-12f);
      U[0][2] = cx / nn; U[1][2] = cy / nn; U[2][2] = cz / nn;
    }
    float Rr[3][3];
    for (int i = 0; i < 3; ++i)
      for (int j = 0; j < 3; ++j) {
        float acc = 0.f;
        for (int k = 0; k < 3; ++k) acc += V[i][k] * U[j][k];
        Rr[i][j] = acc;
      }
    float sg = (det3(Rr) >= 0.f) ? 1.f : -1.f;
    for (int i = 0; i < 3; ++i) V[i][2] *= sg;
    float R[3][3];
    for (int i = 0; i < 3; ++i)
      for (int j = 0; j < 3; ++j) {
        float acc = 0.f;
        for (int k = 0; k < 3; ++k) acc += V[i][k] * U[j][k];
        R[i][j] = acc;
        R_out[b * 9 + i * 3 + j] = acc;
      }
    for (int i = 0; i < 3; ++i) {
      float acc = 0.f;
      for (int j = 0; j < 3; ++j) acc += R[i][j] * sb[j];
      trans_out[b * 3 + i] = gb[i] - acc;
    }
    float tr = R[0][0] + R[1][1] + R[2][2];
    float q[4];
    if (tr > 0.f) {
      float s0 = sqrtf(fmaxf(tr + 1.f, 1e-10f)) * 2.f;
      q[0] = (R[2][1] - R[1][2]) / s0; q[1] = (R[0][2] - R[2][0]) / s0;
      q[2] = (R[1][0] - R[0][1]) / s0; q[3] = 0.25f * s0;
    } else if (R[0][0] > R[1][1] && R[0][0] > R[2][2]) {
      float s1 = sqrtf(fmaxf(1.f + R[0][0] - R[1][1] - R[2][2], 1e-10f)) * 2.f;
      q[0] = 0.25f * s1; q[1] = (R[0][1] + R[1][0]) / s1;
      q[2] = (R[0][2] + R[2][0]) / s1; q[3] = (R[2][1] - R[1][2]) / s1;
    } else if (R[1][1] > R[2][2]) {
      float s2 = sqrtf(fmaxf(1.f + R[1][1] - R[0][0] - R[2][2], 1e-10f)) * 2.f;
      q[0] = (R[0][1] + R[1][0]) / s2; q[1] = 0.25f * s2;
      q[2] = (R[1][2] + R[2][1]) / s2; q[3] = (R[0][2] - R[2][0]) / s2;
    } else {
      float s3 = sqrtf(fmaxf(1.f + R[2][2] - R[0][0] - R[1][1], 1e-10f)) * 2.f;
      q[0] = (R[0][2] + R[2][0]) / s3; q[1] = (R[1][2] + R[2][1]) / s3;
      q[2] = 0.25f * s3; q[3] = (R[1][0] - R[0][1]) / s3;
    }
    float qn = fmaxf(sqrtf(q[0]*q[0] + q[1]*q[1] + q[2]*q[2] + q[3]*q[3]), 1e-12f);
    for (int i = 0; i < 4; ++i) quat_out[b * 4 + i] = q[i] / qn;
  }
}

extern "C" void kernel_launch(void* const* d_in, const int* in_sizes, int n_in,
                              void* d_out, int out_size, void* d_ws, size_t ws_size,
                              hipStream_t stream) {
  const float* Lbat = (const float*)d_in[0];
  const float* srcC = (const float*)d_in[1];
  const float* tgtC = (const float*)d_in[2];
  const float* cw1  = (const float*)d_in[3];
  const float* cb1  = (const float*)d_in[4];
  const float* cw2  = (const float*)d_in[5];
  const float* cb2  = (const float*)d_in[6];
  const float* cw3  = (const float*)d_in[7];
  const float* cb3  = (const float*)d_in[8];
  const float* ow1  = (const float*)d_in[9];
  const float* ob1  = (const float*)d_in[10];
  const float* ow2  = (const float*)d_in[11];
  const float* ob2  = (const float*)d_in[12];
  (void)in_sizes; (void)n_in; (void)out_size; (void)ws_size;

  float* out   = (float*)d_out;
  float* quat  = out;                 // [4,4]
  float* trans = out + 16;            // [4,3]
  float* Rm    = out + 28;            // [4,3,3]
  float* corr  = out + 64;            // [4,256,256]
  float* ovl   = out + 64 + 4 * 256 * 256; // [4,256,1]

  char* ws = (char*)d_ws;
  u16*   W1bf    = (u16*)ws;                                   // 544*256*2
  u16*   W2bf    = (u16*)(ws + 278528);                        // 256*128*2
  float* matched = (float*)(ws + 278528 + 65536);              // 4*256*524*4
  float* tgtm    = (float*)(ws + 278528 + 65536 + 4u*256u*524u*4u);

  conv_w1_kernel<<<(KP * 256 + 255) / 256, 256, 0, stream>>>(cw1, W1bf);
  conv_w2_kernel<<<(256 * 128 + 255) / 256, 256, 0, stream>>>(cw2, W2bf);
  corr_kernel<<<4 * 256, 256, 0, stream>>>(Lbat, tgtC, cb1, cb2, cw3, cb3,
                                           W1bf, W2bf, corr, matched, tgtm);
  overlap_kernel<<<4 * 256, 128, 0, stream>>>(matched, ow1, ob1, ow2, ob2, ovl);
  pose_kernel<<<4, 256, 0, stream>>>(srcC, tgtm, ovl, quat, trans, Rm);
}
